// ExpertMLP_37589553774756
// MI455X (gfx1250) — matrix operand order repaired and checked by reading, not compiled, not run on hardware
//
#include <hip/hip_runtime.h>

// ---------------------------------------------------------------------------
// MoE expert MLP (top-2 of 8), MI455X / gfx1250, wave32 + WMMA bf16.
//   y[s,:] = sum_e c[s,e] * ( silu(x[s,:] @ w1[e]^T) @ w2[e]^T )
// bf16 convert (weights L2-resident) -> deterministic routing/compaction ->
// grouped gather-GEMM1 (+silu,+combine wt) -> grouped GEMM2 with f32 atomic
// scatter (<=2 commutative adds/element => bit-deterministic).
// GEMMs: 64x128 block tile, 8 waves, each wave a 2x2 grid of 16x16x32 bf16
// WMMAs; QUAD-buffered GLOBAL_LOAD_ASYNC_TO_LDS_B128 DMA, one barrier per
// K-step. Fragment loads are inline-asm ds_load_b128 followed by an
// s_wait_dscnt 0x0 that ties the loaded registers, so the whole per-step
// ordering (ACP -> s_wait_asynccnt -> barrier -> ds_load -> wait -> WMMA)
// is enforced by asm-volatile ordering + dataflow. Workspace ~112 MB.
// ---------------------------------------------------------------------------

#define S_TOK 4096
#define DM    1024
#define DF    2048
#define NE    8
#define CAP   8192   // max assignments one expert can receive (S*K)

typedef __bf16 bf16_t;
typedef bf16_t v16bf __attribute__((ext_vector_type(16)));
typedef float  v8f   __attribute__((ext_vector_type(8)));
typedef unsigned int v4u __attribute__((ext_vector_type(4)));

union FragB16 { v16bf v; v4u q[2]; };

__device__ __forceinline__ unsigned short f2bf(float f) {
  unsigned u = __float_as_uint(f);
  u += 0x7FFFu + ((u >> 16) & 1u);          // round-to-nearest-even
  return (unsigned short)(u >> 16);
}

__device__ __forceinline__ unsigned lds_off(const void* p) {
  return (unsigned)(uintptr_t)p;            // low 32 bits = LDS offset
}

// Async DMA global -> LDS, 16 bytes, per-lane addresses, literal byte offset.
#define ACP(ldsv, gp, BOFF)                                                   \
  asm volatile("global_load_async_to_lds_b128 %0, %1, off offset:" BOFF      \
               :: "v"(ldsv), "v"((unsigned long long)(uintptr_t)(gp))        \
               : "memory")

// LDS fragment load, literal byte offset off a 32-bit LDS base address.
#define DSLD(dstq, base, OFF)                                                 \
  asm volatile("ds_load_b128 %0, %1 offset:" OFF : "=v"(dstq) : "v"(base))

// Wait for all asm ds_loads; ties the 8 fragment quads so the WMMAs that
// consume them are dataflow-ordered after this wait.
#define DSWAIT()                                                              \
  asm volatile("s_wait_dscnt 0x0"                                             \
               : "+v"(fa0.q[0]), "+v"(fa0.q[1]), "+v"(fa1.q[0]),              \
                 "+v"(fa1.q[1]), "+v"(fb0.q[0]), "+v"(fb0.q[1]),              \
                 "+v"(fb1.q[0]), "+v"(fb1.q[1]))

// One K-step (32 of K). MODE: 0 = issue batch into buffer Q (byte offsets
// KB/KB16 off gAn/gBn) + wait<=6, 1 = wait<=3, 2 = wait<=0.
// AO0..AO3 / BO0..BO3: literal LDS byte offsets of this step's fragment
// chunks (buffer index folded in). All args are literals.
#define KSTEP(Q, KB, KB16, MODE, AO0, AO1, AO2, AO3, BO0, BO1, BO2, BO3)      \
  do {                                                                        \
    if ((MODE) == 0) {                                                        \
      ACP(ldsA[Q], gAn, KB);                                                  \
      ACP(ldsB[Q], gBn, KB);                                                  \
      ACP(ldsB[Q] + 16, gBn, KB16);                                           \
      asm volatile("s_wait_asynccnt 0x6" ::: "memory");                       \
    } else if ((MODE) == 1) {                                                 \
      asm volatile("s_wait_asynccnt 0x3" ::: "memory");                       \
    } else {                                                                  \
      asm volatile("s_wait_asynccnt 0x0" ::: "memory");                       \
    }                                                                         \
    __syncthreads();                                                          \
    FragB16 fa0, fa1, fb0, fb1;                                               \
    DSLD(fa0.q[0], apA, AO0);  DSLD(fa0.q[1], apA, AO1);                      \
    DSLD(fa1.q[0], apA, AO2);  DSLD(fa1.q[1], apA, AO3);                      \
    DSLD(fb0.q[0], bpA, BO0);  DSLD(fb0.q[1], bpA, BO1);                      \
    DSLD(fb1.q[0], bpA, BO2);  DSLD(fb1.q[1], bpA, BO3);                      \
    DSWAIT();                                                                 \
    a00 = __builtin_amdgcn_wmma_f32_16x16x32_bf16(false, fa0.v, false, fb0.v, (short)0, a00, false, false); \
    a01 = __builtin_amdgcn_wmma_f32_16x16x32_bf16(false, fa0.v, false, fb1.v, (short)0, a01, false, false); \
    a10 = __builtin_amdgcn_wmma_f32_16x16x32_bf16(false, fa1.v, false, fb0.v, (short)0, a10, false, false); \
    a11 = __builtin_amdgcn_wmma_f32_16x16x32_bf16(false, fa1.v, false, fb1.v, (short)0, a11, false, false); \
  } while (0)

// Per-buffer fragment offset literals:
//   A: P*5120 + {0, 32, 1280, 1312}    B: P*10240 + {0, 32, 1280, 1312}
#define KSTEP_P0(Q, KB, KB16, MODE) \
  KSTEP(Q, KB, KB16, MODE, "0","32","1280","1312", "0","32","1280","1312")
#define KSTEP_P1(Q, KB, KB16, MODE) \
  KSTEP(Q, KB, KB16, MODE, "5120","5152","6400","6432", "10240","10272","11520","11552")
#define KSTEP_P2(Q, KB, KB16, MODE) \
  KSTEP(Q, KB, KB16, MODE, "10240","10272","11520","11552", "20480","20512","21760","21792")
#define KSTEP_P3(Q, KB, KB16, MODE) \
  KSTEP(Q, KB, KB16, MODE, "15360","15392","16640","16672", "30720","30752","32000","32032")

// ---------------- fp32 -> bf16 conversion (vectorized, streaming) ----------
__global__ __launch_bounds__(256) void moe_cvt_bf16(const float* __restrict__ src,
                                                    unsigned short* __restrict__ dst,
                                                    int n4) {
  int i = blockIdx.x * 256 + threadIdx.x;
  if (i >= n4) return;
  float4 v = ((const float4*)src)[i];
  uint2 o;
  o.x = (unsigned)f2bf(v.x) | ((unsigned)f2bf(v.y) << 16);
  o.y = (unsigned)f2bf(v.z) | ((unsigned)f2bf(v.w) << 16);
  ((uint2*)dst)[i] = o;
}

// ---------------- routing: deterministic per-expert compaction -------------
__global__ __launch_bounds__(32) void moe_route(const int* __restrict__ tke,
                                                const float* __restrict__ tkw,
                                                int* __restrict__ tok,
                                                float* __restrict__ tw,
                                                int* __restrict__ cntPad) {
  const int e = blockIdx.x;
  const int lane = threadIdx.x;
  int count = 0;
  for (int base = 0; base < S_TOK; base += 32) {
    int s = base + lane;
    int e0 = tke[s * 2 + 0];
    int e1 = tke[s * 2 + 1];
    float w = 0.0f;
    if (e0 == e) w += tkw[s * 2 + 0];
    if (e1 == e) w += tkw[s * 2 + 1];
    bool m = (e0 == e) || (e1 == e);
    unsigned mask = (unsigned)__ballot(m);          // wave32: low 32 bits
    int pre = __popc(mask & ((1u << lane) - 1u));
    if (m) {
      tok[e * CAP + count + pre] = s;
      tw[e * CAP + count + pre] = w;
    }
    count += __popc(mask);
  }
  int padded = (count + 63) & ~63;                  // pad to 64-row M tile
  for (int r = count + lane; r < padded; r += 32) { // zero-weight sentinels
    tok[e * CAP + r] = 0;
    tw[e * CAP + r] = 0.0f;
  }
  if (lane == 0) cntPad[e] = padded;
}

__global__ void moe_prefix(const int* __restrict__ cntPad, int* __restrict__ hoff) {
  if (threadIdx.x == 0 && blockIdx.x == 0) {
    int a = 0;
    for (int e = 0; e < NE; ++e) { hoff[e] = a; a += cntPad[e]; }
  }
}

// ---------------- GEMM1: h = silu(Xg @ w1^T) * c,  bf16 out ----------------
__global__ __launch_bounds__(256) void moe_gemm1(const unsigned short* __restrict__ xb,
                                                 const unsigned short* __restrict__ w1b,
                                                 const int* __restrict__ tok,
                                                 const float* __restrict__ tw,
                                                 const int* __restrict__ cntPad,
                                                 const int* __restrict__ hoff,
                                                 unsigned short* __restrict__ hbuf) {
  const int e = blockIdx.z, mt = blockIdx.x, nt = blockIdx.y;
  if (mt * 64 >= cntPad[e]) return;                 // uniform early exit

  __shared__ unsigned short lA[4][64 * 40];         // 64x32 bf16 per buffer
  __shared__ unsigned short lB[4][128 * 40];        // 128x32 bf16 per buffer
  __shared__ int   sTok[64];
  __shared__ float sW[64];

  const int t = threadIdx.x, lane = t & 31, wv = t >> 5;
  const int wm = wv & 1, wn = wv >> 1;              // 2 x 4 wave grid
  if (t < 64) { sTok[t] = tok[e * CAP + mt * 64 + t]; sW[t] = tw[e * CAP + mt * 64 + t]; }
  __syncthreads();

  const int ar = t >> 2, ac = (t & 3) * 8;          // A loader: 64 rows x 16B
  const int br = t >> 1, bc = (t & 1) * 16;         // B loader: 128 rows x 32B
  const unsigned short* gA = xb + (size_t)sTok[ar] * DM + ac;
  const unsigned short* gB = w1b + ((size_t)e * DF + (size_t)nt * 128 + br) * DM + bc;
  unsigned ldsA[4], ldsB[4];
#pragma unroll
  for (int q = 0; q < 4; ++q) {
    ldsA[q] = lds_off(&lA[q][ar * 40 + ac]);
    ldsB[q] = lds_off(&lB[q][br * 40 + bc]);
  }
  // fragment LDS base addresses (buffer 0); buffer P folded into DS offsets
  const int arow = wm * 32 + (lane & 15);
  const int aklo = (lane < 16) ? 0 : 8;
  const int bk   = (lane < 16) ? 0 : 8;
  const unsigned apA = lds_off(&lA[0][arow * 40 + aklo]);
  const unsigned bpA = lds_off(&lB[0][(wn * 32 + (lane & 15)) * 40 + bk]);

  v8f a00 = {}, a01 = {}, a10 = {}, a11 = {};
  const int nIter = DM / 32;                        // 32 (>=8, %4==0)
  // prologue: batches 0 and 1 in flight
  ACP(ldsA[0], gA, "0");  ACP(ldsB[0], gB, "0");  ACP(ldsB[0] + 16, gB, "16");
  ACP(ldsA[1], gA, "64"); ACP(ldsB[1], gB, "64"); ACP(ldsB[1] + 16, gB, "80");
  const unsigned short* gAn = gA + 64;              // batch it+2 base
  const unsigned short* gBn = gB + 64;

#pragma unroll 1
  for (int it = 0; it < nIter - 4; it += 4) {       // steady state
    KSTEP_P0(2, "0",   "16",  0);
    KSTEP_P1(3, "64",  "80",  0);
    KSTEP_P2(0, "128", "144", 0);
    KSTEP_P3(1, "192", "208", 0);
    gAn += 128; gBn += 128;
  }
  KSTEP_P0(2, "0",  "16", 0);                       // tail (4 iterations)
  KSTEP_P1(3, "64", "80", 0);
  KSTEP_P2(0, "0",  "0",  1);
  KSTEP_P3(0, "0",  "0",  2);

  // C layout: VGPR i -> M = i (+8 for lanes 16-31), N = lane&15.
  const int hb = hoff[e];
  const int n0 = nt * 128 + wn * 32 + (lane & 15);
  const int mbase = wm * 32 + ((lane >> 4) << 3);
#pragma unroll
  for (int i = 0; i < 8; ++i) {
    const int m0 = mbase + i, m1 = m0 + 16;
    const float cw0 = sW[m0], cw1 = sW[m1];
    const size_t r0 = (size_t)(hb + mt * 64 + m0) * DF;
    const size_t r1 = (size_t)(hb + mt * 64 + m1) * DF;
    float v;
    v = a00[i]; hbuf[r0 + n0]      = f2bf(v * cw0 * (1.0f / (1.0f + __expf(-v))));
    v = a01[i]; hbuf[r0 + n0 + 16] = f2bf(v * cw0 * (1.0f / (1.0f + __expf(-v))));
    v = a10[i]; hbuf[r1 + n0]      = f2bf(v * cw1 * (1.0f / (1.0f + __expf(-v))));
    v = a11[i]; hbuf[r1 + n0 + 16] = f2bf(v * cw1 * (1.0f / (1.0f + __expf(-v))));
  }
}

// ---------------- GEMM2: y[token,:] += H @ w2^T  (f32 atomic scatter) ------
__global__ __launch_bounds__(256) void moe_gemm2(const unsigned short* __restrict__ hbuf,
                                                 const unsigned short* __restrict__ w2b,
                                                 const int* __restrict__ tok,
                                                 const int* __restrict__ cntPad,
                                                 const int* __restrict__ hoff,
                                                 float* __restrict__ y) {
  const int e = blockIdx.z, mt = blockIdx.x, nt = blockIdx.y;
  if (mt * 64 >= cntPad[e]) return;

  __shared__ unsigned short lA[4][64 * 40];
  __shared__ unsigned short lB[4][128 * 40];
  __shared__ int sTok[64];

  const int t = threadIdx.x, lane = t & 31, wv = t >> 5;
  const int wm = wv & 1, wn = wv >> 1;
  if (t < 64) sTok[t] = tok[e * CAP + mt * 64 + t];
  __syncthreads();

  const int hb = hoff[e];
  const int ar = t >> 2, ac = (t & 3) * 8;
  const int br = t >> 1, bc = (t & 1) * 16;
  const unsigned short* gA = hbuf + (size_t)(hb + mt * 64 + ar) * DF + ac;
  const unsigned short* gB = w2b + ((size_t)e * DM + (size_t)nt * 128 + br) * DF + bc;
  unsigned ldsA[4], ldsB[4];
#pragma unroll
  for (int q = 0; q < 4; ++q) {
    ldsA[q] = lds_off(&lA[q][ar * 40 + ac]);
    ldsB[q] = lds_off(&lB[q][br * 40 + bc]);
  }
  const int arow = wm * 32 + (lane & 15);
  const int aklo = (lane < 16) ? 0 : 8;
  const int bk   = (lane < 16) ? 0 : 8;
  const unsigned apA = lds_off(&lA[0][arow * 40 + aklo]);
  const unsigned bpA = lds_off(&lB[0][(wn * 32 + (lane & 15)) * 40 + bk]);

  v8f a00 = {}, a01 = {}, a10 = {}, a11 = {};
  const int nIter = DF / 32;                        // 64 (>=8, %4==0)
  ACP(ldsA[0], gA, "0");  ACP(ldsB[0], gB, "0");  ACP(ldsB[0] + 16, gB, "16");
  ACP(ldsA[1], gA, "64"); ACP(ldsB[1], gB, "64"); ACP(ldsB[1] + 16, gB, "80");
  const unsigned short* gAn = gA + 64;
  const unsigned short* gBn = gB + 64;

#pragma unroll 1
  for (int it = 0; it < nIter - 4; it += 4) {
    KSTEP_P0(2, "0",   "16",  0);
    KSTEP_P1(3, "64",  "80",  0);
    KSTEP_P2(0, "128", "144", 0);
    KSTEP_P3(1, "192", "208", 0);
    gAn += 128; gBn += 128;
  }
  KSTEP_P0(2, "0",  "16", 0);
  KSTEP_P1(3, "64", "80", 0);
  KSTEP_P2(0, "0",  "0",  1);
  KSTEP_P3(0, "0",  "0",  2);

  const int n0 = nt * 128 + wn * 32 + (lane & 15);
  const int mbase = wm * 32 + ((lane >> 4) << 3);
#pragma unroll
  for (int i = 0; i < 8; ++i) {
    const int m0 = mbase + i, m1 = m0 + 16;
    const int s0 = sTok[m0], s1 = sTok[m1];         // padded rows add +0.0
    atomicAdd(&y[(size_t)s0 * DM + n0],      a00[i]);
    atomicAdd(&y[(size_t)s0 * DM + n0 + 16], a01[i]);
    atomicAdd(&y[(size_t)s1 * DM + n0],      a10[i]);
    atomicAdd(&y[(size_t)s1 * DM + n0 + 16], a11[i]);
  }
}

// ---------------- workspace layout (bytes) ---------------------------------
static const size_t OFF_W1B  = 0;                          // 8*2048*1024*2
static const size_t OFF_W2B  = OFF_W1B + 33554432;         // 8*1024*2048*2
static const size_t OFF_XB   = OFF_W2B + 33554432;         // 4096*1024*2
static const size_t OFF_TOK  = OFF_XB  + 8388608;          // 8*8192*4
static const size_t OFF_TW   = OFF_TOK + 262144;           // 8*8192*4
static const size_t OFF_CNT  = OFF_TW  + 262144;           // 8*4 (padded 64)
static const size_t OFF_HOFF = OFF_CNT + 64;               // 8*4 (padded 64)
static const size_t OFF_H    = OFF_HOFF + 64;              // <=8704 rows *2048*2
// total ~ 111.7 MB

extern "C" void kernel_launch(void* const* d_in, const int* in_sizes, int n_in,
                              void* d_out, int out_size, void* d_ws, size_t ws_size,
                              hipStream_t stream) {
  const float* x   = (const float*)d_in[0];
  const int*   tke = (const int*)d_in[1];
  const float* tkw = (const float*)d_in[2];
  const float* w1  = (const float*)d_in[3];
  const float* w2  = (const float*)d_in[4];
  float* y = (float*)d_out;

  char* ws = (char*)d_ws;
  unsigned short* w1b  = (unsigned short*)(ws + OFF_W1B);
  unsigned short* w2b  = (unsigned short*)(ws + OFF_W2B);
  unsigned short* xb   = (unsigned short*)(ws + OFF_XB);
  int*            tokI = (int*)(ws + OFF_TOK);
  float*          tokW = (float*)(ws + OFF_TW);
  int*            cnt  = (int*)(ws + OFF_CNT);
  int*            hoff = (int*)(ws + OFF_HOFF);
  unsigned short* hbuf = (unsigned short*)(ws + OFF_H);

  hipMemsetAsync(y, 0, (size_t)out_size * sizeof(float), stream);

  const int nW = NE * DF * DM;          // 16,777,216 (w1 and w2 each)
  const int nX = S_TOK * DM;            // 4,194,304
  moe_cvt_bf16<<<nW / 4 / 256, 256, 0, stream>>>(w1, w1b, nW / 4);
  moe_cvt_bf16<<<nW / 4 / 256, 256, 0, stream>>>(w2, w2b, nW / 4);
  moe_cvt_bf16<<<nX / 4 / 256, 256, 0, stream>>>(x, xb, nX / 4);

  moe_route<<<NE, 32, 0, stream>>>(tke, tkw, tokI, tokW, cnt);
  moe_prefix<<<1, 1, 0, stream>>>(cnt, hoff);

  // Oversized static grids; blocks past the routed count exit immediately.
  moe_gemm1<<<dim3(CAP / 64, DF / 128, NE), 256, 0, stream>>>(xb, w1b, tokI, tokW,
                                                              cnt, hoff, hbuf);
  moe_gemm2<<<dim3(CAP / 64, DM / 128, NE), 256, 0, stream>>>(hbuf, w2b, tokI,
                                                              cnt, hoff, y);
}
